// PPO_51058571215432
// MI455X (gfx1250) — compile-verified, matrix operand order checked by reading, not run on hardware
//
#include <hip/hip_runtime.h>
#include <hip/hip_fp16.h>

typedef _Float16 v16h __attribute__((ext_vector_type(16)));
typedef _Float16 v8h  __attribute__((ext_vector_type(8)));
typedef float    v8f  __attribute__((ext_vector_type(8)));

#define NN        30000
#define MM        16
#define ROWS      (NN * MM)      // 480000
#define KCONV     261
#define KCONV_PAD 288
#define BN_EPS    1e-5f

// ---------------------------------------------------------------- WMMA helpers

__device__ __forceinline__ v8f wmma_f32_16x16x32_f16(v16h a, v16h b, v8f c) {
  // (neg_a, A, neg_b, B, c_mod, C, reuse_a, reuse_b)
  return __builtin_amdgcn_wmma_f32_16x16x32_f16(false, a, false, b, (short)0, c,
                                                false, false);
}

// Build a 16x32 f16 A fragment from an LDS tile (row-major, `stride` halves).
// CDNA5 layout: lanes 0-15 hold rows 0-15 with K = {k0..k0+7, k0+16..k0+23};
// lanes 16-31 hold rows 0-15 with K = {k0+8..k0+15, k0+24..k0+31}.
__device__ __forceinline__ v16h load_a_frag(const _Float16* base, int stride,
                                            int lane, int k0) {
  const int row  = lane & 15;
  const int koff = (lane & 16) ? 8 : 0;
  const v8h lo = *(const v8h*)(base + row * stride + k0 + koff);
  const v8h hi = *(const v8h*)(base + row * stride + k0 + 16 + koff);
  return __builtin_shufflevector(lo, hi, 0, 1, 2, 3, 4, 5, 6, 7,
                                 8, 9, 10, 11, 12, 13, 14, 15);
}

// float <-> order-preserving unsigned (for atomicMax on floats incl. negatives)
__device__ __forceinline__ unsigned fenc(float x) {
  int i = __float_as_int(x);
  return (i >= 0) ? ((unsigned)i | 0x80000000u) : ~(unsigned)i;
}
__device__ __forceinline__ float fdec(unsigned u) {
  int i = (u & 0x80000000u) ? (int)(u & 0x7fffffffu) : (int)~u;
  return __int_as_float(i);
}

// ---------------------------------------------------------------- init / pack

__global__ void k_init(float* stats, int nstats, unsigned* gmax, float* gsum) {
  for (int i = blockIdx.x * blockDim.x + threadIdx.x; i < nstats;
       i += gridDim.x * blockDim.x)
    stats[i] = 0.f;
  if (blockIdx.x == 0 && threadIdx.x == 0) { *gmax = 0u; *gsum = 0.f; }
}

// Pack fp32 weights W[Ksrc x ldw] (using cols 0..Ncols-1) into the per-lane
// WMMA B-fragment order: dst[((ct*S + s)*32 + lane)*16 + j], mirroring the
// A-fragment K grouping so each lane reads its fragment contiguously (32B).
__global__ void k_pack(const float* __restrict__ W, int ldw, int Ksrc, int Kpad,
                       _Float16* __restrict__ dst, int total) {
  const int S = Kpad >> 5;
  for (int i = blockIdx.x * blockDim.x + threadIdx.x; i < total;
       i += gridDim.x * blockDim.x) {
    int j    = i & 15;
    int lane = (i >> 4) & 31;
    int rest = i >> 9;
    int s    = rest % S;
    int ct   = rest / S;
    int kg   = (j & 8) ? 16 : 0;
    int koff = (lane & 16) ? 8 : 0;
    int k    = s * 32 + kg + koff + (j & 7);
    int c    = ct * 16 + (lane & 15);
    float v  = (k < Ksrc) ? W[(size_t)k * ldw + c] : 0.f;
    dst[i]   = (_Float16)v;
  }
}

// ---------------------------------------------------------------- embedding

__global__ void k_embed(const float* __restrict__ nf, const float* __restrict__ W,
                        const float* __restrict__ b, float* __restrict__ node_f,
                        _Float16* __restrict__ node_h) {
  int i = blockIdx.x * blockDim.x + threadIdx.x;
  int n = i >> 7, c = i & 127;
  float acc = b[c];
#pragma unroll
  for (int k = 0; k < 19; ++k) acc += nf[n * 19 + k] * W[k * 128 + c];
  node_f[i] = acc;
  node_h[i] = (_Float16)acc;
}

// ---------------------------------------------------------------- conv GEMM
// One block per node n (its 16 neighbor-rows). 8 waves each own a 16x16 output
// tile (128 filter cols). Second gated half is dead code in the reference
// (split + discard, BN is per-feature) so it is never computed.

__global__ void __launch_bounds__(256)
k_conv(const _Float16* __restrict__ node_h, const int* __restrict__ idx,
       const float* __restrict__ edge, const _Float16* __restrict__ wp,
       const float* __restrict__ bias, _Float16* __restrict__ gated,
       float* __restrict__ s1, float* __restrict__ q1) {
  __shared__ __align__(32) _Float16 As[16 * KCONV_PAD];
  const int n   = blockIdx.x;
  const int tid = threadIdx.x;

  // Stage A tile: [self(128) | gathered nbr(128) | edge(5) | pad(27)] x 16 rows
  for (int e = tid; e < 16 * KCONV_PAD; e += 256) {
    int r = e / KCONV_PAD, c = e - r * KCONV_PAD;
    _Float16 v;
    if (c < 128) {
      v = node_h[(size_t)n * 128 + c];
    } else if (c < 256) {
      int gi = idx[n * 16 + r];
      if (gi < 0) gi += NN;  // torch-style negative-index wrap
      v = node_h[(size_t)gi * 128 + (c - 128)];
    } else if (c < 261) {
      v = (_Float16)edge[((size_t)n * 16 + r) * 5 + (c - 256)];
    } else {
      v = (_Float16)0.f;
    }
    As[r * KCONV_PAD + c] = v;
  }
  __syncthreads();

  const int wave = tid >> 5, lane = tid & 31;
  v8f acc = {};
#pragma unroll
  for (int s = 0; s < 9; ++s) {
    v16h a = load_a_frag(As, KCONV_PAD, lane, s * 32);
    v16h b = *(const v16h*)(wp + ((size_t)(wave * 9 + s) * 32 + lane) * 16);
    acc = wmma_f32_16x16x32_f16(a, b, acc);
  }

  // C layout: lane&15 = col; lanes<16 rows 0-7, lanes>=16 rows 8-15.
  const int col = wave * 16 + (lane & 15);
  const float bb = bias[col];
  const int r0 = (lane & 16) ? 8 : 0;
  float s = 0.f, q = 0.f;
#pragma unroll
  for (int r = 0; r < 8; ++r) {
    float v = acc[r] + bb;
    s += v;
    q += v * v;
    gated[((size_t)n * 16 + (r0 + r)) * 128 + col] = (_Float16)v;
  }
  // Fold the two half-tiles (same col, different rows) then one atomic per col.
  s += __shfl_xor(s, 16, 32);
  q += __shfl_xor(q, 16, 32);
  if (lane < 16) {
    atomicAdd(&s1[col], s);
    atomicAdd(&q1[col], q);
  }
}

// ---------------------------------------------------------------- BN finalize

__global__ void k_bnfin(const float* __restrict__ sum, const float* __restrict__ sq,
                        float* __restrict__ mu, float* __restrict__ rs, float cnt) {
  int c = threadIdx.x;
  float m = sum[c] / cnt;
  float v = sq[c] / cnt - m * m;  // biased variance (training-mode BN)
  mu[c] = m;
  rs[c] = rsqrtf(v + BN_EPS);
}

// ------------------------------------------- BN1 apply + gate + neighbor-sum

__global__ void __launch_bounds__(256)
k_reduce(const _Float16* __restrict__ gated, const int* __restrict__ idx,
         const float* __restrict__ mu1, const float* __restrict__ rs1,
         const float* __restrict__ g1, const float* __restrict__ b1,
         float* __restrict__ summed, float* __restrict__ s2,
         float* __restrict__ q2) {
  const int t = threadIdx.x;
  const int c = t & 127, sub = t >> 7;
  const int n = blockIdx.x * 2 + sub;
  const float mu = mu1[c], rs = rs1[c], gg = g1[c], bb = b1[c];
  float acc = 0.f;
#pragma unroll
  for (int m = 0; m < 16; ++m) {
    float gv = (float)gated[((size_t)n * 16 + m) * 128 + c];
    float x = (gv - mu) * rs * gg + bb;
    float sg = 1.f / (1.f + expf(-x));
    float msk = (idx[n * 16 + m] >= 0) ? 1.f : 0.f;
    acc += sg * sg * msk;  // faithful to the reference's filt*core bug
  }
  summed[(size_t)n * 128 + c] = acc;
  __shared__ float r1[256], r2[256];
  r1[t] = acc;
  r2[t] = acc * acc;
  __syncthreads();
  if (t < 128) {
    atomicAdd(&s2[c], r1[t] + r1[t + 128]);
    atomicAdd(&q2[c], r2[t] + r2[t + 128]);
  }
}

// ---------------------------------------------------------------- node update

__global__ void k_update(float* __restrict__ node_f, _Float16* __restrict__ node_h,
                         const float* __restrict__ summed,
                         const float* __restrict__ mu2, const float* __restrict__ rs2,
                         const float* __restrict__ g2, const float* __restrict__ b2) {
  int i = blockIdx.x * blockDim.x + threadIdx.x;
  int c = i & 127;
  float x = node_f[i] + (summed[i] - mu2[c]) * rs2[c] * g2[c] + b2[c];
  float y = (x > 20.f) ? x : log1pf(expf(x));  // softplus
  node_f[i] = y;
  node_h[i] = (_Float16)y;
}

// ---------------------------------------------------------------- PI MLP
// 4 waves per block, each wave owns one 16-row tile (= one node's neighbors)
// and runs the full 148->128->256->128->1 chain in LDS with WMMA.

__global__ void __launch_bounds__(128)
k_pi(const _Float16* __restrict__ node_h, const int* __restrict__ idx,
     const float* __restrict__ edge, const float* __restrict__ dist,
     const int* __restrict__ tp, const _Float16* __restrict__ wp1,
     const float* __restrict__ b1, const _Float16* __restrict__ wp2,
     const float* __restrict__ b2, const _Float16* __restrict__ wp3,
     const float* __restrict__ b3, const float* __restrict__ w4,
     const float* __restrict__ b4, float* __restrict__ logits,
     unsigned* __restrict__ gmax) {
  __shared__ __align__(32) _Float16 sA[4][16 * 256];
  __shared__ __align__(32) _Float16 sB[4][16 * 128];
  const int w = threadIdx.x >> 5, lane = threadIdx.x & 31;
  const int tile = blockIdx.x * 4 + w;  // == node n (M == 16)
  const float tpf = (float)tp[0];
  _Float16* Abuf = sA[w];
  _Float16* Bbuf = sB[w];

  // feat tile 16 x 160 (148 real + 12 zero pad)
  for (int e = lane; e < 16 * 160; e += 32) {
    int r = e / 160, c = e - r * 160;
    int grow = tile * 16 + r;
    _Float16 v;
    if (c < 128) {
      int gi = idx[grow];
      if (gi < 0) gi += NN;
      v = node_h[(size_t)gi * 128 + c];
    } else if (c < 138) {
      v = (_Float16)edge[(size_t)grow * 5 + ((c - 128) % 5)];  // edge tiled x2
    } else if (c < 143) {
      v = (_Float16)dist[grow];  // distance repeated x5
    } else if (c < 148) {
      v = (_Float16)tpf;  // tp_type x5
    } else {
      v = (_Float16)0.f;
    }
    Abuf[r * 160 + c] = v;
  }
  __syncthreads();

  // h1 = relu(feat @ W1 + b1): 16x128, K=160 (5 steps) -> Bbuf
  for (int ct = 0; ct < 8; ++ct) {
    v8f acc = {};
#pragma unroll
    for (int s = 0; s < 5; ++s) {
      v16h a = load_a_frag(Abuf, 160, lane, s * 32);
      v16h bf = *(const v16h*)(wp1 + ((size_t)(ct * 5 + s) * 32 + lane) * 16);
      acc = wmma_f32_16x16x32_f16(a, bf, acc);
    }
    int col = ct * 16 + (lane & 15);
    float bb = b1[col];
    int r0 = (lane & 16) ? 8 : 0;
#pragma unroll
    for (int r = 0; r < 8; ++r) {
      float v = acc[r] + bb;
      v = v > 0.f ? v : 0.f;
      Bbuf[(r0 + r) * 128 + col] = (_Float16)v;
    }
  }
  __syncthreads();

  // h2 = relu(h1 @ W2 + b2): 16x256, K=128 (4 steps) -> Abuf
  for (int ct = 0; ct < 16; ++ct) {
    v8f acc = {};
#pragma unroll
    for (int s = 0; s < 4; ++s) {
      v16h a = load_a_frag(Bbuf, 128, lane, s * 32);
      v16h bf = *(const v16h*)(wp2 + ((size_t)(ct * 4 + s) * 32 + lane) * 16);
      acc = wmma_f32_16x16x32_f16(a, bf, acc);
    }
    int col = ct * 16 + (lane & 15);
    float bb = b2[col];
    int r0 = (lane & 16) ? 8 : 0;
#pragma unroll
    for (int r = 0; r < 8; ++r) {
      float v = acc[r] + bb;
      v = v > 0.f ? v : 0.f;
      Abuf[(r0 + r) * 256 + col] = (_Float16)v;
    }
  }
  __syncthreads();

  // h3 = relu(h2 @ W3 + b3): 16x128, K=256 (8 steps) -> Bbuf
  for (int ct = 0; ct < 8; ++ct) {
    v8f acc = {};
#pragma unroll
    for (int s = 0; s < 8; ++s) {
      v16h a = load_a_frag(Abuf, 256, lane, s * 32);
      v16h bf = *(const v16h*)(wp3 + ((size_t)(ct * 8 + s) * 32 + lane) * 16);
      acc = wmma_f32_16x16x32_f16(a, bf, acc);
    }
    int col = ct * 16 + (lane & 15);
    float bb = b3[col];
    int r0 = (lane & 16) ? 8 : 0;
#pragma unroll
    for (int r = 0; r < 8; ++r) {
      float v = acc[r] + bb;
      v = v > 0.f ? v : 0.f;
      Bbuf[(r0 + r) * 128 + col] = (_Float16)v;
    }
  }
  __syncthreads();

  // logits (N=1 column: cheaper as a VALU dot) + action mask + global max
  float lg = -3.0e38f;
  if (lane < 16) {
    int grow = tile * 16 + lane;
    float acc = b4[0];
    for (int k = 0; k < 128; ++k)
      acc += (float)Bbuf[lane * 128 + k] * w4[k];
    bool valid = (idx[grow] >= 0) && (edge[(size_t)grow * 5 + 4] <= tpf);
    if (!valid) acc -= 1e8f;
    logits[grow] = acc;
    lg = acc;
  }
#pragma unroll
  for (int d = 1; d < 32; d <<= 1) lg = fmaxf(lg, __shfl_xor(lg, d, 32));
  if (lane == 0) atomicMax(gmax, fenc(lg));
}

// ---------------------------------------------------------------- softmax

__global__ void __launch_bounds__(256)
k_expsum(const float* __restrict__ logits, const unsigned* __restrict__ gmax,
         float* __restrict__ gsum) {
  int i = blockIdx.x * 256 + threadIdx.x;
  float mx = fdec(*gmax);
  float e = expf(logits[i] - mx);
#pragma unroll
  for (int d = 1; d < 32; d <<= 1) e += __shfl_xor(e, d, 32);
  __shared__ float red[8];
  int lane = threadIdx.x & 31, wv = threadIdx.x >> 5;
  if (lane == 0) red[wv] = e;
  __syncthreads();
  if (threadIdx.x == 0) {
    float s = 0.f;
    for (int k = 0; k < 8; ++k) s += red[k];
    atomicAdd(gsum, s);
  }
}

__global__ void k_norm(const float* __restrict__ logits,
                       const unsigned* __restrict__ gmax,
                       const float* __restrict__ gsum, float* __restrict__ out) {
  int i = blockIdx.x * blockDim.x + threadIdx.x;
  float mx = fdec(*gmax);
  out[i] = expf(logits[i] - mx) / (*gsum);
}

// ---------------------------------------------------------------- launch

extern "C" void kernel_launch(void* const* d_in, const int* in_sizes, int n_in,
                              void* d_out, int out_size, void* d_ws,
                              size_t ws_size, hipStream_t stream) {
  (void)in_sizes; (void)n_in; (void)out_size; (void)ws_size;
  const float* node_fea = (const float*)d_in[0];
  const float* edge_fea = (const float*)d_in[1];
  const int*   efidx    = (const int*)d_in[2];
  const float* distance = (const float*)d_in[3];
  const int*   tp_type  = (const int*)d_in[4];
  const float* emb_W    = (const float*)d_in[5];
  const float* emb_b    = (const float*)d_in[6];
  const float* conv_W   = (const float*)d_in[7];
  const float* conv_b   = (const float*)d_in[8];
  const float* bn1g     = (const float*)d_in[9];
  const float* bn1b     = (const float*)d_in[10];
  const float* bn2g     = (const float*)d_in[11];
  const float* bn2b     = (const float*)d_in[12];
  const float* piW1     = (const float*)d_in[13];
  const float* pib1     = (const float*)d_in[14];
  const float* piW2     = (const float*)d_in[15];
  const float* pib2     = (const float*)d_in[16];
  const float* piW3     = (const float*)d_in[17];
  const float* pib3     = (const float*)d_in[18];
  const float* piW4     = (const float*)d_in[19];
  const float* pib4     = (const float*)d_in[20];

  char* ws = (char*)d_ws;
  size_t off = 0;
  auto take = [&](size_t bytes) {
    size_t o = off;
    off += (bytes + 255) & ~(size_t)255;
    return o;
  };
  float*    node_f = (float*)(ws + take((size_t)NN * 128 * 4));
  _Float16* node_h = (_Float16*)(ws + take((size_t)NN * 128 * 2));
  _Float16* gated  = (_Float16*)(ws + take((size_t)NN * 16 * 128 * 2));
  float*    summed = (float*)(ws + take((size_t)NN * 128 * 4));
  _Float16* wpc    = (_Float16*)(ws + take((size_t)3 * KCONV_PAD * 128 * 2));
  _Float16* wp1    = (_Float16*)(ws + take((size_t)160 * 128 * 2));
  _Float16* wp2    = (_Float16*)(ws + take((size_t)128 * 256 * 2));
  _Float16* wp3    = (_Float16*)(ws + take((size_t)256 * 128 * 2));
  float*    stats  = (float*)(ws + take((size_t)8 * 3 * 128 * 4));
  float*    logits = (float*)(ws + take((size_t)ROWS * 4));
  char*     gred   = ws + take(256);
  unsigned* gmax   = (unsigned*)gred;
  float*    gsum   = (float*)(gred + 4);

  // stats slots: 0:s1sum 1:s1sq 2:s1mu 3:s1rs 4:s2sum 5:s2sq 6:s2mu 7:s2rs
  auto ST = [&](int which, int l) { return stats + ((size_t)which * 3 + l) * 128; };

  k_init<<<16, 256, 0, stream>>>(stats, 8 * 3 * 128, gmax, gsum);

  for (int l = 0; l < 3; ++l) {
    int total = KCONV_PAD * 128;
    k_pack<<<(total + 255) / 256, 256, 0, stream>>>(
        conv_W + (size_t)l * 261 * 256, 256, 261, KCONV_PAD,
        wpc + (size_t)l * KCONV_PAD * 128, total);
  }
  k_pack<<<(160 * 128 + 255) / 256, 256, 0, stream>>>(piW1, 128, 148, 160, wp1, 160 * 128);
  k_pack<<<(128 * 256 + 255) / 256, 256, 0, stream>>>(piW2, 256, 128, 128, wp2, 128 * 256);
  k_pack<<<(256 * 128 + 255) / 256, 256, 0, stream>>>(piW3, 128, 256, 256, wp3, 256 * 128);

  k_embed<<<NN * 128 / 256, 256, 0, stream>>>(node_fea, emb_W, emb_b, node_f, node_h);

  for (int l = 0; l < 3; ++l) {
    k_conv<<<NN, 256, 0, stream>>>(node_h, efidx, edge_fea,
                                   wpc + (size_t)l * KCONV_PAD * 128,
                                   conv_b + l * 256, gated, ST(0, l), ST(1, l));
    k_bnfin<<<1, 128, 0, stream>>>(ST(0, l), ST(1, l), ST(2, l), ST(3, l), (float)ROWS);
    k_reduce<<<NN / 2, 256, 0, stream>>>(gated, efidx, ST(2, l), ST(3, l),
                                         bn1g + l * 256, bn1b + l * 256, summed,
                                         ST(4, l), ST(5, l));
    k_bnfin<<<1, 128, 0, stream>>>(ST(4, l), ST(5, l), ST(6, l), ST(7, l), (float)NN);
    k_update<<<NN * 128 / 256, 256, 0, stream>>>(node_f, node_h, summed, ST(6, l),
                                                 ST(7, l), bn2g + l * 128, bn2b + l * 128);
  }

  k_pi<<<NN / 4, 128, 0, stream>>>(node_h, efidx, edge_fea, distance, tp_type, wp1,
                                   pib1, wp2, pib2, wp3, pib3, piW4, pib4, logits, gmax);
  k_expsum<<<ROWS / 256, 256, 0, stream>>>(logits, gmax, gsum);
  k_norm<<<ROWS / 256, 256, 0, stream>>>(logits, gmax, gsum, (float*)d_out);
}
